// LSTM_AE_1108101562864
// MI455X (gfx1250) — matrix operand order repaired and checked by reading, not compiled, not run on hardware
//
#include <hip/hip_runtime.h>
#include <hip/hip_bf16.h>

typedef _Float16 half8   __attribute__((ext_vector_type(8)));
typedef _Float16 half16  __attribute__((ext_vector_type(16)));
typedef float    floatx2 __attribute__((ext_vector_type(2)));
typedef float    floatx8 __attribute__((ext_vector_type(8)));

#define B_   128
#define T_   256
#define H_   512
#define IN_  64
#define G4_  2048      // 4*H
#define HT_  131072    // H*T
#define ENCN (B_ * H_ * T_)   // 16777216 floats of "encoded" output
#define FC_CHUNKS 64
#define FC_KC (HT_ / FC_CHUNKS)   // 2048

// ---------------------------------------------------------------------------
// small prep kernels
// ---------------------------------------------------------------------------
__global__ void k_cvt_f16(const float* __restrict__ s, _Float16* __restrict__ d, int n) {
    int i = blockIdx.x * blockDim.x + threadIdx.x;
    if (i < n) d[i] = (_Float16)s[i];
}
__global__ void k_copy_f32(const float* __restrict__ s, float* __restrict__ d, int n) {
    int i = blockIdx.x * blockDim.x + threadIdx.x;
    if (i < n) d[i] = s[i];
}
__global__ void k_bias_combine(const float* __restrict__ a, const float* __restrict__ b,
                               float* __restrict__ d, int n) {
    int i = blockIdx.x * blockDim.x + threadIdx.x;
    if (i < n) d[i] = a[i] + b[i];
}
// x: (B, IN, T) f32  ->  x16: (T, B, IN) f16
__global__ void k_xpose(const float* __restrict__ x, _Float16* __restrict__ x16) {
    int i = blockIdx.x * blockDim.x + threadIdx.x;   // i = ((t*B)+b)*IN + k
    if (i >= T_ * B_ * IN_) return;
    int t = i >> 13;               // /(B_*IN_) = /8192
    int r = i & 8191;
    int b = r >> 6;
    int k = r & 63;
    x16[i] = (_Float16)x[(size_t)(b * IN_ + k) * T_ + t];
}

// ---------------------------------------------------------------------------
// WMMA f16 GEMM fragment helper (16x16x32, f32 accumulate), software-pipelined.
//   gates[b0:+16, h0:+16 (+g*512)] += A[b0:+16, :K] * W[g*512+h0:+16, :K]^T
// A-frag (16x32 f16): lane m=lane&15; halfs 0-7 = K[kk+ad..+8), 8-15 = K[kk+8+ad..+8)
// B-frag (32x16 f16): lane n=lane&15 selects row of W; 16 contiguous halfs at kk+bd
// ---------------------------------------------------------------------------
__device__ __forceinline__ void gates_gemm(floatx8* acc,
        const _Float16* __restrict__ A, int lda,
        const _Float16* __restrict__ W, int ldw, int K,
        int b0, int h0, int lane)
{
    const int ad = (lane & 16) ? 16 : 0;
    const int bd = (lane & 16) ? 16 : 0;
    const _Float16* arow = A + (size_t)(b0 + (lane & 15)) * lda;
    const _Float16* wrow = W + (size_t)(h0 + (lane & 15)) * ldw + bd;
    const size_t gstride = (size_t)H_ * ldw;

    // prologue: fragments for kk = 0
    half8 lo = *(const half8*)(arow + ad);
    half8 hi = *(const half8*)(arow + 8 + ad);
    half16 a_cur = __builtin_shufflevector(lo, hi,
                    0,1,2,3,4,5,6,7,8,9,10,11,12,13,14,15);
    half16 b_cur[4];
#pragma unroll
    for (int g = 0; g < 4; ++g)
        b_cur[g] = *(const half16*)(wrow + g * gstride);

#pragma unroll
    for (int kk = 0; kk < K; kk += 32) {
        half16 a_nxt = a_cur;
        half16 b_nxt[4] = { b_cur[0], b_cur[1], b_cur[2], b_cur[3] };
        const int kn = kk + 32;
        if (kn < K) {   // prefetch next iteration's fragments before the WMMAs
            half8 l2 = *(const half8*)(arow + kn + ad);
            half8 h2 = *(const half8*)(arow + kn + 8 + ad);
            a_nxt = __builtin_shufflevector(l2, h2,
                        0,1,2,3,4,5,6,7,8,9,10,11,12,13,14,15);
#pragma unroll
            for (int g = 0; g < 4; ++g)
                b_nxt[g] = *(const half16*)(wrow + kn + g * gstride);
        }
#pragma unroll
        for (int g = 0; g < 4; ++g)
            acc[g] = __builtin_amdgcn_wmma_f32_16x16x32_f16(
                         false, a_cur, false, b_cur[g], (short)0, acc[g], false, false);
        a_cur = a_nxt;
#pragma unroll
        for (int g = 0; g < 4; ++g) b_cur[g] = b_nxt[g];
    }
}

// C/D frag: lane L, reg r -> row b = b0 + r + (L&16 ? 8:0), col h = h0 + (L&15)
__device__ __forceinline__ void lstm_pointwise(floatx8* acc,
        const float* __restrict__ bias, float* __restrict__ c_state,
        _Float16* __restrict__ h_next, _Float16* __restrict__ h_copy,
        float* __restrict__ f32out, int tcol, int b0, int h0, int lane)
{
    const int h = h0 + (lane & 15);
    const float bi = bias[h];
    const float bf = bias[H_ + h];
    const float bg = bias[2 * H_ + h];
    const float bo = bias[3 * H_ + h];
    const int rb = b0 + ((lane & 16) ? 8 : 0);
#pragma unroll
    for (int r = 0; r < 8; ++r) {
        int b = rb + r;
        float gi = acc[0][r] + bi;
        float gf = acc[1][r] + bf;
        float gg = acc[2][r] + bg;
        float go = acc[3][r] + bo;
        float si = 1.0f / (1.0f + __expf(-gi));
        float sf = 1.0f / (1.0f + __expf(-gf));
        float so = 1.0f / (1.0f + __expf(-go));
        float c  = sf * c_state[b * H_ + h] + si * tanhf(gg);
        c_state[b * H_ + h] = c;
        float hv = so * tanhf(c);
        h_next[b * H_ + h] = (_Float16)hv;
        if (h_copy) h_copy[b * H_ + h] = (_Float16)hv;
        // f32out laid out as (B, H*T): [b, h*T + tcol]
        f32out[(size_t)b * HT_ + (size_t)h * T_ + tcol] = hv;
    }
}

// ---------------------------------------------------------------------------
// fused step kernel: blocks 0..31 = encoder step t, blocks 32..63 = decoder step t-1
// grid 64 x 256 threads (8 waves). wave tile: 16(batch) x 16(hidden) x 4 gates.
// h states double-buffered by step parity (no intra-launch RAW hazard).
// ---------------------------------------------------------------------------
__global__ void k_lstm_step(int t,
        const _Float16* __restrict__ x16,
        const _Float16* __restrict__ WihE, const _Float16* __restrict__ WhhE,
        const float* __restrict__ bE,
        _Float16* __restrict__ hE16, float* __restrict__ cE,
        _Float16* __restrict__ hsE16, float* __restrict__ enc_out,
        const _Float16* __restrict__ WihD, const _Float16* __restrict__ WhhD,
        const float* __restrict__ bD,
        _Float16* __restrict__ hD16, float* __restrict__ cD,
        float* __restrict__ hsD)
{
    const int lane = threadIdx.x & 31;
    const int widx = threadIdx.x >> 5;
    const bool is_dec = (blockIdx.x >= 32);
    if (!is_dec && t >= T_) return;
    if (is_dec && t < 1) return;

    const int wid = ((blockIdx.x & 31) << 3) + widx;   // 0..255
    const int b0 = (wid >> 5) * 16;                    // 8 M-tiles
    const int h0 = (wid & 31) * 16;                    // 32 N-tiles

    floatx8 acc[4] = {};

    if (!is_dec) {
        const _Float16* h_read  = hE16 + (size_t)(t & 1) * (B_ * H_);
        _Float16*       h_write = hE16 + (size_t)((t + 1) & 1) * (B_ * H_);
        gates_gemm(acc, x16 + (size_t)t * (B_ * IN_), IN_, WihE, IN_, IN_, b0, h0, lane);
        gates_gemm(acc, h_read, H_, WhhE, H_, H_, b0, h0, lane);
        lstm_pointwise(acc, bE, cE, h_write,
                       hsE16 + (size_t)t * (B_ * H_), enc_out, t, b0, h0, lane);
    } else {
        const int td = t - 1;
        const _Float16* h_read  = hD16 + (size_t)(td & 1) * (B_ * H_);
        _Float16*       h_write = hD16 + (size_t)((td + 1) & 1) * (B_ * H_);
        gates_gemm(acc, hsE16 + (size_t)td * (B_ * H_), H_, WihD, H_, H_, b0, h0, lane);
        gates_gemm(acc, h_read, H_, WhhD, H_, H_, b0, h0, lane);
        lstm_pointwise(acc, bD, cD, h_write, nullptr, hsD, td, b0, h0, lane);
    }
}

// ---------------------------------------------------------------------------
// FC stage 1: partial[chunk][b][o] = sum over K-chunk of hsD[b,k]*Wfc[o,k]
// fp32 WMMA 16x16x4; tiles 8(M) x 32(N) = 256; 64 K-chunks of 2048.
// 16384 waves = 2048 blocks of 8 waves. Deterministic (no atomics).
// ---------------------------------------------------------------------------
__global__ void k_fc(const float* __restrict__ hsD, const float* __restrict__ Wfc,
                     float* __restrict__ partial)
{
    const int lane = threadIdx.x & 31;
    const int gid = blockIdx.x * 8 + (threadIdx.x >> 5);  // 0..16383
    const int tile  = gid >> 6;
    const int chunk = gid & 63;
    const int b0 = (tile >> 5) * 16;
    const int o0 = (tile & 31) * 16;
    const int m  = lane & 15;
    const int koff = (lane & 16) ? 2 : 0;

    const float* arow = hsD + (size_t)(b0 + m) * HT_ + koff;
    const float* brow = Wfc + (size_t)(o0 + m) * HT_ + koff;

    floatx8 acc = {0.f,0.f,0.f,0.f,0.f,0.f,0.f,0.f};
    const int kb = chunk * FC_KC;
    for (int k = kb; k < kb + FC_KC; k += 16) {
        floatx2 a[4], b[4];
#pragma unroll
        for (int u = 0; u < 4; ++u) {
            a[u] = *(const floatx2*)(arow + k + 4 * u);
            b[u] = *(const floatx2*)(brow + k + 4 * u);
        }
#pragma unroll
        for (int u = 0; u < 4; ++u)
            acc = __builtin_amdgcn_wmma_f32_16x16x4_f32(
                      false, a[u], false, b[u], (short)0, acc, false, false);
    }
    const int rb = b0 + ((lane & 16) ? 8 : 0);
    float* p = partial + (size_t)chunk * (B_ * H_);
#pragma unroll
    for (int r = 0; r < 8; ++r)
        p[(size_t)(rb + r) * H_ + o0 + (lane & 15)] = acc[r];
}

// FC stage 2: fixed-order reduction over chunks + bias (deterministic)
__global__ void k_fc_reduce(const float* __restrict__ partial,
                            const float* __restrict__ bfc,
                            float* __restrict__ dec)
{
    int i = blockIdx.x * blockDim.x + threadIdx.x;
    if (i >= B_ * H_) return;
    float s = bfc[i & (H_ - 1)];
#pragma unroll 8
    for (int c = 0; c < FC_CHUNKS; ++c)
        s += partial[(size_t)c * (B_ * H_) + i];
    dec[i] = s;
}

// ---------------------------------------------------------------------------
// host launcher
// ---------------------------------------------------------------------------
static inline int cdiv(int a, int b) { return (a + b - 1) / b; }

extern "C" void kernel_launch(void* const* d_in, const int* in_sizes, int n_in,
                              void* d_out, int out_size, void* d_ws, size_t ws_size,
                              hipStream_t stream) {
    (void)in_sizes; (void)n_in; (void)out_size; (void)ws_size;
    const float* x     = (const float*)d_in[0];
    const float* h0e   = (const float*)d_in[1];
    const float* c0e   = (const float*)d_in[2];
    const float* WihE  = (const float*)d_in[3];
    const float* WhhE  = (const float*)d_in[4];
    const float* bihE  = (const float*)d_in[5];
    const float* bhhE  = (const float*)d_in[6];
    const float* h0d   = (const float*)d_in[7];
    const float* c0d   = (const float*)d_in[8];
    const float* WihD  = (const float*)d_in[9];
    const float* WhhD  = (const float*)d_in[10];
    const float* bihD  = (const float*)d_in[11];
    const float* bhhD  = (const float*)d_in[12];
    const float* Wfc   = (const float*)d_in[13];
    const float* bfc   = (const float*)d_in[14];

    float* enc_out = (float*)d_out;
    float* dec_out = enc_out + (size_t)ENCN;

    // workspace layout (256B aligned segments)
    char* base = (char*)d_ws;
    size_t off = 0;
    auto alloc = [&](size_t bytes) -> char* {
        char* p = base + off;
        off += (bytes + 255) & ~(size_t)255;
        return p;
    };
    _Float16* wWihE = (_Float16*)alloc((size_t)G4_ * IN_ * 2);
    _Float16* wWhhE = (_Float16*)alloc((size_t)G4_ * H_ * 2);
    _Float16* wWihD = (_Float16*)alloc((size_t)G4_ * H_ * 2);
    _Float16* wWhhD = (_Float16*)alloc((size_t)G4_ * H_ * 2);
    float*    wbE   = (float*)   alloc((size_t)G4_ * 4);
    float*    wbD   = (float*)   alloc((size_t)G4_ * 4);
    _Float16* x16   = (_Float16*)alloc((size_t)T_ * B_ * IN_ * 2);
    _Float16* hE16  = (_Float16*)alloc((size_t)2 * B_ * H_ * 2);   // double buffer
    float*    cE    = (float*)   alloc((size_t)B_ * H_ * 4);
    _Float16* hD16  = (_Float16*)alloc((size_t)2 * B_ * H_ * 2);   // double buffer
    float*    cD    = (float*)   alloc((size_t)B_ * H_ * 4);
    _Float16* hsE16 = (_Float16*)alloc((size_t)T_ * B_ * H_ * 2);
    float*    hsD   = (float*)   alloc((size_t)B_ * HT_ * 4);
    float*    fcP   = (float*)   alloc((size_t)FC_CHUNKS * B_ * H_ * 4);

    const int TB = 256;
    // weight/bias/state prep
    k_cvt_f16<<<cdiv(G4_ * IN_, TB), TB, 0, stream>>>(WihE, wWihE, G4_ * IN_);
    k_cvt_f16<<<cdiv(G4_ * H_, TB), TB, 0, stream>>>(WhhE, wWhhE, G4_ * H_);
    k_cvt_f16<<<cdiv(G4_ * H_, TB), TB, 0, stream>>>(WihD, wWihD, G4_ * H_);
    k_cvt_f16<<<cdiv(G4_ * H_, TB), TB, 0, stream>>>(WhhD, wWhhD, G4_ * H_);
    k_bias_combine<<<cdiv(G4_, TB), TB, 0, stream>>>(bihE, bhhE, wbE, G4_);
    k_bias_combine<<<cdiv(G4_, TB), TB, 0, stream>>>(bihD, bhhD, wbD, G4_);
    k_xpose<<<cdiv(T_ * B_ * IN_, TB), TB, 0, stream>>>(x, x16);
    k_cvt_f16<<<cdiv(B_ * H_, TB), TB, 0, stream>>>(h0e, hE16, B_ * H_); // parity-0 buffer
    k_cvt_f16<<<cdiv(B_ * H_, TB), TB, 0, stream>>>(h0d, hD16, B_ * H_);
    k_copy_f32<<<cdiv(B_ * H_, TB), TB, 0, stream>>>(c0e, cE, B_ * H_);
    k_copy_f32<<<cdiv(B_ * H_, TB), TB, 0, stream>>>(c0d, cD, B_ * H_);

    // fused encoder(t) + decoder(t-1) recurrence; t=T_ runs only decoder's last step
    for (int t = 0; t <= T_; ++t) {
        k_lstm_step<<<dim3(64), dim3(256), 0, stream>>>(t,
            x16, wWihE, wWhhE, wbE, hE16, cE, hsE16, enc_out,
            wWihD, wWhhD, wbD, hD16, cD, hsD);
    }

    // final FC: split-K fp32 WMMA partials, then deterministic reduction
    k_fc<<<dim3(2048), dim3(256), 0, stream>>>(hsD, Wfc, fcP);
    k_fc_reduce<<<dim3(cdiv(B_ * H_, TB)), dim3(TB), 0, stream>>>(fcP, bfc, dec_out);
}
